// DeformableConvolution1d_44109314130518
// MI455X (gfx1250) — matrix operand order repaired and checked by reading, not compiled, not run on hardware
//
#include <hip/hip_runtime.h>

typedef __attribute__((ext_vector_type(16))) __bf16 v16bf;
typedef __attribute__((ext_vector_type(8)))  float  v8f;

#define Bn    8
#define Cin   64
#define Cout  64
#define Ldim  16384
#define Kk    5
#define PADn  2
#define Lp    (Ldim + 2 * PADn)
#define CK    (Cin * Kk)          // 320
#define CKP   336                 // padded row stride (bf16 elems): 672B, 32B-aligned
#define TILE  128                 // output positions per workgroup
#define NKCH  10                  // K chunks of 32: 320/32
#define NTPW  4                   // N tiles per wave

__device__ __forceinline__ int reflect_src(int ip) {
    // map padded index ip in [0, Lp-1] -> source index with reflect padding
    int idx = ip - PADn;
    idx = (idx < 0) ? -idx : idx;
    idx = (idx > Ldim - 1) ? (2 * (Ldim - 1) - idx) : idx;
    return idx;
}

__global__ __launch_bounds__(256)
void deform_conv1d_wmma_kernel(const float* __restrict__ x,
                               const float* __restrict__ offsets,
                               const float* __restrict__ weight,
                               const float* __restrict__ bias,
                               float* __restrict__ out) {
    __shared__ int    sIdx0[Kk * TILE];
    __shared__ int    sIdx1[Kk * TILE];
    __shared__ float  sFrac[Kk * TILE];
    __shared__ __bf16 sV[TILE][CKP];      // V^T: row = position n, col = ck = c*Kk + k

    const int tid       = threadIdx.x;
    const int b         = blockIdx.y;
    const int tile_base = blockIdx.x * TILE;

    // ---- Phase 1: per-(n,k) sampling positions (channel-independent) ----
    for (int idx = tid; idx < Kk * TILE; idx += 256) {
        int n  = idx / Kk;
        int k  = idx - n * Kk;
        int ng = tile_base + n;
        float off = offsets[(b * Ldim + ng) * Kk + k];
        float T   = (float)(ng + k) + off;
        T = fminf(fmaxf(T, 0.0f), (float)(Lp - 1));
        int i0 = (int)floorf(T);
        i0 = (i0 < 0) ? 0 : ((i0 > Lp - 2) ? (Lp - 2) : i0);
        float fr = T - (float)i0;
        sIdx0[k * TILE + n] = reflect_src(i0);
        sIdx1[k * TILE + n] = reflect_src(i0 + 1);
        sFrac[k * TILE + n] = fr;
    }
    __syncthreads();

    // ---- Phase 2: build V^T tile in LDS as bf16 ----
    {
        const int n  = tid & (TILE - 1);   // position within tile (0..127)
        const int cg = tid >> 7;           // channel group 0/1 (32 channels each)
        int   i0r[Kk], i1r[Kk];
        float frr[Kk];
#pragma unroll
        for (int k = 0; k < Kk; ++k) {
            i0r[k] = sIdx0[k * TILE + n];
            i1r[k] = sIdx1[k * TILE + n];
            frr[k] = sFrac[k * TILE + n];
        }
        const float* xb = x + (size_t)b * Cin * Ldim;
        for (int cc = 0; cc < 32; ++cc) {
            int c = cg * 32 + cc;
            const float* xr = xb + (size_t)c * Ldim;
#pragma unroll
            for (int k = 0; k < Kk; ++k) {
                float g0 = xr[i0r[k]];
                float g1 = xr[i1r[k]];
                float v  = g0 + (g1 - g0) * frr[k];
                sV[n][c * Kk + k] = (__bf16)v;
            }
        }
    }

    // ---- Phase 3: load A fragments (weights, 16 rows x 320 K) per wave ----
    const int w    = tid >> 5;
    const int lane = tid & 31;
    const int mt   = w & 3;            // M tile (16 output channels)
    const int nt0  = (w >> 2) * NTPW;  // first of 4 N tiles for this wave
    const int mrow = lane & 15;
    const int h    = lane >> 4;        // K-half selector

    const float* wrow = weight + (size_t)(mt * 16 + mrow) * CK;
    __builtin_prefetch(wrow, 0, 3);                 // -> global_prefetch_b8
    __builtin_prefetch(wrow + CK - 1, 0, 3);

    v16bf afrag[NKCH];
#pragma unroll
    for (int kk = 0; kk < NKCH; ++kk) {
        int kb = kk * 32 + h * 16;
#pragma unroll
        for (int j = 0; j < 8; ++j) {
            afrag[kk][j]     = (__bf16)wrow[kb + j];       // K = kk*32 + h*16 + j
            afrag[kk][j + 8] = (__bf16)wrow[kb + 8 + j];   // K = kk*32 + h*16 + 8 + j
        }
    }
    __syncthreads();

    // ---- Phase 4: WMMA K-loop over 10 chunks of 32, 4 N tiles per wave ----
    const int ncol = lane & 15;                    // B/D column within N tile
    const int row0 = nt0 * 16 + ncol;              // LDS row (position) for N tile 0

    v8f acc[NTPW];
#pragma unroll
    for (int j = 0; j < NTPW; ++j) acc[j] = (v8f){};

#pragma unroll
    for (int kk = 0; kk < NKCH; ++kk) {
        int ckb = kk * 32 + h * 16;                // lanes 0-15: K 0..15, 16-31: K 16..31
#pragma unroll
        for (int j = 0; j < NTPW; ++j) {
            v16bf bfrag = *(const v16bf*)&sV[row0 + j * 16][ckb];
            acc[j] = __builtin_amdgcn_wmma_f32_16x16x32_bf16(
                         false, afrag[kk], false, bfrag, (short)0, acc[j], false, false);
        }
    }

    // ---- Phase 5: write out (C/D layout: VGPR r -> M = r + 8*h, N = lane&15) ----
    const int nbase = tile_base + nt0 * 16 + ncol;
#pragma unroll
    for (int r = 0; r < 8; ++r) {
        int o = mt * 16 + r + h * 8;
        float bv = bias[o];
        size_t rowoff = ((size_t)b * Cout + o) * Ldim;
#pragma unroll
        for (int j = 0; j < NTPW; ++j) {
            out[rowoff + nbase + j * 16] = acc[j][r] + bv;
        }
    }
}

extern "C" void kernel_launch(void* const* d_in, const int* in_sizes, int n_in,
                              void* d_out, int out_size, void* d_ws, size_t ws_size,
                              hipStream_t stream) {
    const float* x       = (const float*)d_in[0];
    const float* offsets = (const float*)d_in[1];
    const float* weight  = (const float*)d_in[2];
    const float* bias    = (const float*)d_in[3];
    float* out = (float*)d_out;

    dim3 grid(Ldim / TILE, Bn);   // (128, 8)
    dim3 block(256);
    deform_conv1d_wmma_kernel<<<grid, block, 0, stream>>>(x, offsets, weight, bias, out);
}
